// TverskyLayer_37400575213810
// MI455X (gfx1250) — matrix operand order repaired and checked by reading, not compiled, not run on hardware
//
#include <hip/hip_runtime.h>
#include <stdint.h>

// ---------------------------------------------------------------------------
// Types for CDNA5 WMMA (gfx1250, wave32): v_wmma_f32_16x16x32_bf16
// ---------------------------------------------------------------------------
typedef __attribute__((ext_vector_type(16))) __bf16 v16bf;
typedef __attribute__((ext_vector_type(8)))  __bf16 v8bf;
typedef __attribute__((ext_vector_type(8)))  float  v8f;

union FragU { v16bf v; v8bf h[2]; };

__device__ __forceinline__ unsigned short f2bf(float f) {
  unsigned int u = __float_as_uint(f);
  u += 0x7FFFu + ((u >> 16) & 1u);   // round-to-nearest-even
  return (unsigned short)(u >> 16);
}

// A-fragment: 16x32 bf16 tile, row-major source with leading dim `ld` (elems).
// Lane l (m = l&15, h = l>>4) holds row m, K = {h*8..h*8+7} U {16+h*8..16+h*8+7}.
__device__ __forceinline__ v16bf ldA(const unsigned short* tile, int ld, int lane, int k0) {
  const int m = lane & 15, h = lane >> 4;
  const unsigned short* p = tile + (size_t)m * ld + (size_t)(k0 + h * 8);
  FragU u;
  u.h[0] = *(const v8bf*)(p);        // K = k0 + h*8 .. +7     (global_load_b128)
  u.h[1] = *(const v8bf*)(p + 16);   // K = k0 + 16 + h*8 .. +7
  return u.v;
}

// B-fragment: 32x16 bf16 (held as rows of W, i.e. B^T row-major, K contiguous).
// Lane l (n = l&15, h = l>>4) holds column n, K = {h*8..h*8+7} U {16+h*8..16+h*8+7}.
__device__ __forceinline__ v16bf ldB(const unsigned short* tile, int ld, int lane, int k0) {
  const int n = lane & 15, h = lane >> 4;
  const unsigned short* p = tile + (size_t)n * ld + (size_t)(k0 + h * 8);
  FragU u;
  u.h[0] = *(const v8bf*)(p);
  u.h[1] = *(const v8bf*)(p + 16);
  return u.v;
}

__device__ __forceinline__ v8f wmma_bf16(v16bf a, v16bf b, v8f c) {
  // (neg_a, A, neg_b, B, c_mod, C, reuse_a, reuse_b)
  return __builtin_amdgcn_wmma_f32_16x16x32_bf16(false, a, false, b, (short)0, c,
                                                 false, false);
}

// ---------------------------------------------------------------------------
// FP32 -> BF16 conversion (vectorized: float4 in, 4x bf16 out)
// ---------------------------------------------------------------------------
struct alignas(8) US4 { unsigned short x, y, z, w; };

__global__ void k_cvt_bf16(const float4* __restrict__ src, US4* __restrict__ dst, int n4) {
  int i = blockIdx.x * blockDim.x + threadIdx.x;
  if (i < n4) {
    float4 f = src[i];
    US4 o;
    o.x = f2bf(f.x); o.y = f2bf(f.y); o.z = f2bf(f.z); o.w = f2bf(f.w);
    dst[i] = o;
  }
}

// ---------------------------------------------------------------------------
// Stage 1: Out = A @ features.T (K = 1024), fused Tversky epilogue.
//   v      = (A @ F.T)[row,col]
//   pres   = max(v, 0)
//   main   = v * pres                        (this is `a` or `c`)
//   aux    = t0*main + t1*(1 - pres)         (this is `V` or `U`)
// mode 0 (prototypes): t0 = theta, t1 = -alpha  -> main=c, aux=U
// mode 1 (x):          t0 = 0,     t1 = -beta   -> main=a, aux=V
// Block = 128 threads = 4 waves (2x2), wave tile = 64x64, block tile = 128x128.
// ---------------------------------------------------------------------------
__global__ void __launch_bounds__(128)
k_stage1(const unsigned short* __restrict__ Abf,
         const unsigned short* __restrict__ Fbf,
         unsigned short* __restrict__ outMain,
         unsigned short* __restrict__ outAux,
         const float* __restrict__ s_theta,
         const float* __restrict__ s_ab,
         int mode) {
  const int lane = threadIdx.x & 31;
  const int wave = threadIdx.x >> 5;
  const int R = blockIdx.y * 128 + (wave >> 1) * 64;   // output row base
  const int C = blockIdx.x * 128 + (wave & 1) * 64;    // output col base
  const unsigned short* Ab = Abf + (size_t)R * 1024;
  const unsigned short* Bb = Fbf + (size_t)C * 1024;

  const v8f vzero = {};
  v8f acc[4][4];
#pragma unroll
  for (int i = 0; i < 4; ++i)
#pragma unroll
    for (int j = 0; j < 4; ++j) acc[i][j] = vzero;

  for (int k0 = 0; k0 < 1024; k0 += 32) {
    v16bf a[4], b[4];
#pragma unroll
    for (int i = 0; i < 4; ++i) a[i] = ldA(Ab + (size_t)(16 * i) * 1024, 1024, lane, k0);
#pragma unroll
    for (int j = 0; j < 4; ++j) b[j] = ldB(Bb + (size_t)(16 * j) * 1024, 1024, lane, k0);
    if (k0 + 32 < 1024) {  // hint next K-slab into cache (global_prefetch_b8)
      __builtin_prefetch(Ab + (size_t)(lane * 2) * 1024 + (size_t)(k0 + 32), 0, 1);
      __builtin_prefetch(Bb + (size_t)(lane * 2) * 1024 + (size_t)(k0 + 32), 0, 1);
    }
#pragma unroll
    for (int i = 0; i < 4; ++i)
#pragma unroll
      for (int j = 0; j < 4; ++j) acc[i][j] = wmma_bf16(a[i], b[j], acc[i][j]);
  }

  const float t0 = (mode == 0) ? s_theta[0] : 0.0f;
  const float t1 = -s_ab[0];
  const int half = lane >> 4, cn = lane & 15;
#pragma unroll
  for (int i = 0; i < 4; ++i)
#pragma unroll
    for (int j = 0; j < 4; ++j) {
#pragma unroll
      for (int r = 0; r < 8; ++r) {
        const int row = R + 16 * i + 8 * half + r;   // C/D layout: vgpr r -> M = r + 8*(lane>=16)
        const int col = C + 16 * j + cn;
        const float v    = acc[i][j][r];
        const float pres = fmaxf(v, 0.0f);
        const float mv   = v * pres;
        const float av   = t0 * mv + t1 * (1.0f - pres);
        const size_t idx = (size_t)row * 2048 + (size_t)col;
        outMain[idx] = f2bf(mv);
        outAux[idx]  = f2bf(av);
      }
    }
}

// ---------------------------------------------------------------------------
// Stage 2: out = a @ U.T + V @ c.T   (M=4096, N=512, K=2048), f32 output.
// ---------------------------------------------------------------------------
__global__ void __launch_bounds__(128)
k_stage2(const unsigned short* __restrict__ abf,
         const unsigned short* __restrict__ Vbf,
         const unsigned short* __restrict__ Ubf,
         const unsigned short* __restrict__ cbf,
         float* __restrict__ out) {
  const int lane = threadIdx.x & 31;
  const int wave = threadIdx.x >> 5;
  const int R = blockIdx.y * 128 + (wave >> 1) * 64;
  const int C = blockIdx.x * 128 + (wave & 1) * 64;
  const unsigned short* A1 = abf + (size_t)R * 2048;
  const unsigned short* A2 = Vbf + (size_t)R * 2048;
  const unsigned short* B1 = Ubf + (size_t)C * 2048;
  const unsigned short* B2 = cbf + (size_t)C * 2048;

  const v8f vzero = {};
  v8f acc[4][4];
#pragma unroll
  for (int i = 0; i < 4; ++i)
#pragma unroll
    for (int j = 0; j < 4; ++j) acc[i][j] = vzero;

  for (int k0 = 0; k0 < 2048; k0 += 32) {
    v16bf a[4], b[4];
    // a @ U.T contribution
#pragma unroll
    for (int i = 0; i < 4; ++i) a[i] = ldA(A1 + (size_t)(16 * i) * 2048, 2048, lane, k0);
#pragma unroll
    for (int j = 0; j < 4; ++j) b[j] = ldB(B1 + (size_t)(16 * j) * 2048, 2048, lane, k0);
#pragma unroll
    for (int i = 0; i < 4; ++i)
#pragma unroll
      for (int j = 0; j < 4; ++j) acc[i][j] = wmma_bf16(a[i], b[j], acc[i][j]);
    // V @ c.T contribution
#pragma unroll
    for (int i = 0; i < 4; ++i) a[i] = ldA(A2 + (size_t)(16 * i) * 2048, 2048, lane, k0);
#pragma unroll
    for (int j = 0; j < 4; ++j) b[j] = ldB(B2 + (size_t)(16 * j) * 2048, 2048, lane, k0);
    if (k0 + 32 < 2048) {
      __builtin_prefetch(A1 + (size_t)(lane * 2) * 2048 + (size_t)(k0 + 32), 0, 1);
      __builtin_prefetch(A2 + (size_t)(lane * 2) * 2048 + (size_t)(k0 + 32), 0, 1);
    }
#pragma unroll
    for (int i = 0; i < 4; ++i)
#pragma unroll
      for (int j = 0; j < 4; ++j) acc[i][j] = wmma_bf16(a[i], b[j], acc[i][j]);
  }

  const int half = lane >> 4, cn = lane & 15;
#pragma unroll
  for (int i = 0; i < 4; ++i)
#pragma unroll
    for (int j = 0; j < 4; ++j) {
#pragma unroll
      for (int r = 0; r < 8; ++r) {
        const int row = R + 16 * i + 8 * half + r;
        const int col = C + 16 * j + cn;
        out[(size_t)row * 512 + (size_t)col] = acc[i][j][r];
      }
    }
}

// ---------------------------------------------------------------------------
// Launch: inputs (x, features, prototypes, alpha, beta, theta) all f32.
// Workspace layout (49 MiB total, fits in MI455X's 192 MB L2):
//   xbf  [4096x1024] @  0 MiB    fbf [2048x1024] @  8 MiB   pbf [512x1024] @ 12 MiB
//   Ubf  [ 512x2048] @ 13 MiB    cbf [ 512x2048] @ 15 MiB
//   abf  [4096x2048] @ 17 MiB    Vbf [4096x2048] @ 33 MiB
// ---------------------------------------------------------------------------
extern "C" void kernel_launch(void* const* d_in, const int* in_sizes, int n_in,
                              void* d_out, int out_size, void* d_ws, size_t ws_size,
                              hipStream_t stream) {
  const float* x     = (const float*)d_in[0];
  const float* feat  = (const float*)d_in[1];
  const float* proto = (const float*)d_in[2];
  const float* alpha = (const float*)d_in[3];
  const float* beta  = (const float*)d_in[4];
  const float* theta = (const float*)d_in[5];
  float* out = (float*)d_out;

  char* ws = (char*)d_ws;
  unsigned short* xbf = (unsigned short*)(ws);
  unsigned short* fbf = (unsigned short*)(ws + ((size_t)8  << 20));
  unsigned short* pbf = (unsigned short*)(ws + ((size_t)12 << 20));
  unsigned short* Ubf = (unsigned short*)(ws + ((size_t)13 << 20));
  unsigned short* cbf = (unsigned short*)(ws + ((size_t)15 << 20));
  unsigned short* abf = (unsigned short*)(ws + ((size_t)17 << 20));
  unsigned short* Vbf = (unsigned short*)(ws + ((size_t)33 << 20));

  // 1) fp32 -> bf16 conversions
  k_cvt_bf16<<<4096, 256, 0, stream>>>((const float4*)x,     (US4*)xbf, 4096 * 1024 / 4);
  k_cvt_bf16<<<2048, 256, 0, stream>>>((const float4*)feat,  (US4*)fbf, 2048 * 1024 / 4);
  k_cvt_bf16<<< 512, 256, 0, stream>>>((const float4*)proto, (US4*)pbf,  512 * 1024 / 4);

  // 2) pf = prototypes @ F.T -> (c, U)   [M=512,  N=2048, K=1024]
  k_stage1<<<dim3(16, 4), 128, 0, stream>>>(pbf, fbf, cbf, Ubf, theta, alpha, 0);

  // 3) xf = x @ F.T -> (a, V)            [M=4096, N=2048, K=1024]
  k_stage1<<<dim3(16, 32), 128, 0, stream>>>(xbf, fbf, abf, Vbf, theta, beta, 1);

  // 4) out = a @ U.T + V @ c.T           [M=4096, N=512, K=2048]
  k_stage2<<<dim3(4, 32), 128, 0, stream>>>(abf, Vbf, Ubf, cbf, out);
}
